// ToxicophoreAttention_86663850098765
// MI455X (gfx1250) — matrix operand order repaired and checked by reading, not compiled, not run on hardware
//
#include <hip/hip_runtime.h>
#include <hip/hip_bf16.h>

#define H     512
#define NH    8
#define HD    64
#define GNUM  4096
#define LN_EPS 1e-5f

typedef __attribute__((ext_vector_type(16))) __bf16 v16bf;
typedef __attribute__((ext_vector_type(8)))  float  v8f;

struct Pack32 { uint4 a, b; };

__device__ inline unsigned short f2bf(float f) {
    unsigned int u = __float_as_uint(f);
    u += 0x7fffu + ((u >> 16) & 1u);          // round-to-nearest-even
    return (unsigned short)(u >> 16);
}

// A-matrix 16x32 bf16 tile load from a row-major bf16 matrix.
// Per ISA 7.12.2: lanes 0-15 hold K {k0..k0+7, k0+16..k0+23},
// lanes 16-31 hold K {k0+8..k0+15, k0+24..k0+31}; M = lane%16.
__device__ inline v16bf load_a16(const unsigned short* base, int row0,
                                 int strideE, int k0, int lane) {
    int m    = lane & 15;
    int koff = k0 + ((lane & 16) ? 8 : 0);
    const uint4* p = (const uint4*)(base + (size_t)(row0 + m) * strideE + koff);
    Pack32 r; r.a = p[0]; r.b = p[2];         // +16 elems = +32 bytes
    return __builtin_bit_cast(v16bf, r);
}

// B-matrix tile from pre-swizzled buffer: [(rt*16+kt)*32 + lane] * 16 elems contiguous.
__device__ inline v16bf load_b16(const unsigned short* bbuf, int rt, int kt, int lane) {
    const uint4* p = (const uint4*)(bbuf + (size_t)((((rt << 4) + kt) << 5) + lane) * 16);
    Pack32 r; r.a = p[0]; r.b = p[1];
    return __builtin_bit_cast(v16bf, r);
}

// ---- Prep: fused score weights  Wscore[h,c] = (1/8) * sum_d toxq[h,d]*Wk[h*64+d, c]
__global__ void k_prep_score(const float* __restrict__ Wk, const float* __restrict__ bk,
                             const float* __restrict__ toxq,
                             float* __restrict__ wscore, float* __restrict__ sbias) {
    int t = blockIdx.x * blockDim.x + threadIdx.x;
    if (t < NH * H) {
        int h = t >> 9, c = t & (H - 1);
        float s = 0.f;
        for (int d = 0; d < HD; ++d)
            s = fmaf(toxq[h * HD + d], Wk[(size_t)(h * HD + d) * H + c], s);
        wscore[t] = s * 0.125f;
    } else if (t < NH * H + NH) {
        int h = t - NH * H;
        float s = 0.f;
        for (int d = 0; d < HD; ++d) s = fmaf(toxq[h * HD + d], bk[h * HD + d], s);
        sbias[h] = s * 0.125f;
    }
}

// ---- Prep: pack Wv and Wo (row-major f32 [512][512]) into bf16 B-matrix swizzle.
// value(ot,kt,lane,e) = W[o = ot*16 + lane%16][k = kt*32 + 8*(lane/16) + (e%8) + 16*(e/8)]
__global__ void k_pack_w(const float* __restrict__ Wv, const float* __restrict__ Wo,
                         unsigned short* __restrict__ wvp, unsigned short* __restrict__ wop) {
    int t = blockIdx.x * blockDim.x + threadIdx.x;   // 0 .. 2*H*H-1
    const float* src = (t < H * H) ? Wv : Wo;
    unsigned short* dst = (t < H * H) ? wvp : wop;
    int i = t & (H * H - 1);
    int e = i & 15, lane = (i >> 4) & 31, kt = (i >> 9) & 15, ot = i >> 13;
    int o = (ot << 4) + (lane & 15);
    int k = (kt << 5) + (e & 7) + ((e & 8) ? 16 : 0) + ((lane & 16) ? 8 : 0);
    dst[i] = f2bf(src[(size_t)o * H + k]);
}

// ---- Prep: graph start offsets via binary search (batch is sorted).
__global__ void k_start(const int* __restrict__ batch, int n, int* __restrict__ start) {
    int g = blockIdx.x * blockDim.x + threadIdx.x;
    if (g > GNUM) return;
    int lo = 0, hi = n;
    while (lo < hi) { int mid = (lo + hi) >> 1; if (batch[mid] < g) lo = mid + 1; else hi = mid; }
    start[g] = lo;
}

// ---- Fused per-graph pass: e = exp(score), denom, xw[h,:] = sum_n e[n,h]*x[n,:],
// then write xwn = xw/denom as bf16 [G][NH*H]. One workgroup per graph, no atomics.
__global__ __launch_bounds__(256) void k_attn(const float* __restrict__ x,
                                              const int* __restrict__ start,
                                              const float* __restrict__ wscore,
                                              const float* __restrict__ sbias,
                                              unsigned short* __restrict__ xwn) {
    __shared__ float sW[NH * H];     // 16 KB fused score weights
    __shared__ float sX[8][H];       // 16 KB staged x rows (chunk of 8 nodes)
    __shared__ float sE[8][NH];
    __shared__ float sDen[NH];
    __shared__ float sSb[NH];

    int tid = threadIdx.x, lane = tid & 31, w = tid >> 5;
    for (int i = tid; i < NH * H; i += 256) sW[i] = wscore[i];
    if (tid < NH) { sSb[tid] = sbias[tid]; sDen[tid] = 0.f; }
    __syncthreads();

    int g = blockIdx.x;
    int s0 = start[g], s1 = start[g + 1];

    float acc[16];                   // lane owns columns c0, c0+32 for 8 heads
    #pragma unroll
    for (int i = 0; i < 16; ++i) acc[i] = 0.f;
    int c0 = w * 64 + lane;

    for (int base = s0; base < s1; base += 8) {
        int cnt = min(8, s1 - base);
        // Phase A: wave w computes e[] for node base+w, stages x row in LDS
        if (w < cnt) {
            int n = base + w;
            const float* xr = x + (size_t)n * H;
            float xv[16];
            #pragma unroll
            for (int j = 0; j < 16; ++j) {
                xv[j] = xr[lane + 32 * j];
                sX[w][lane + 32 * j] = xv[j];
            }
            float e[NH];
            #pragma unroll
            for (int h = 0; h < NH; ++h) {
                float p = 0.f;
                #pragma unroll
                for (int j = 0; j < 16; ++j)
                    p = fmaf(xv[j], sW[h * H + lane + 32 * j], p);
                p += __shfl_xor(p, 16); p += __shfl_xor(p, 8); p += __shfl_xor(p, 4);
                p += __shfl_xor(p, 2);  p += __shfl_xor(p, 1);
                e[h] = __expf(p + sSb[h]);
            }
            if (lane == 0) {
                #pragma unroll
                for (int h = 0; h < NH; ++h) sE[w][h] = e[h];
            }
        }
        __syncthreads();
        // Phase B: all 256 threads accumulate their column pair over the chunk
        for (int i = 0; i < cnt; ++i) {
            float x0 = sX[i][c0], x1 = sX[i][c0 + 32];
            #pragma unroll
            for (int h = 0; h < NH; ++h) {
                float eh = sE[i][h];
                acc[2 * h]     = fmaf(eh, x0, acc[2 * h]);
                acc[2 * h + 1] = fmaf(eh, x1, acc[2 * h + 1]);
            }
        }
        if (tid < NH) {              // deterministic denom accumulation
            float d = sDen[tid];
            for (int i = 0; i < cnt; ++i) d += sE[i][tid];
            sDen[tid] = d;
        }
        __syncthreads();
    }

    unsigned short* dst = xwn + (size_t)g * (NH * H);
    #pragma unroll
    for (int h = 0; h < NH; ++h) {
        float inv = 1.0f / sDen[h];
        dst[h * H + c0]      = f2bf(acc[2 * h] * inv);
        dst[h * H + c0 + 32] = f2bf(acc[2 * h + 1] * inv);
    }
}

// ---- out[g, rt*16+n] = xwn[g, head*512 + :] . Wv[rt*16+n, :] + bv   (head = rt/4)
__global__ __launch_bounds__(32) void k_gemm_v(const unsigned short* __restrict__ xwn,
                                               const unsigned short* __restrict__ wvp,
                                               const float* __restrict__ bv,
                                               unsigned short* __restrict__ outb) {
    int lane = threadIdx.x;
    int gt = blockIdx.x;             // 0..255 (g tiles of 16)
    int rt = blockIdx.y;             // 0..31  (output col tiles of 16)
    int head = rt >> 2;
    v8f acc = {};
    #pragma unroll
    for (int kt = 0; kt < 16; ++kt) {
        v16bf a = load_a16(xwn, gt * 16, NH * H, head * H + kt * 32, lane);
        v16bf b = load_b16(wvp, rt, kt, lane);
        acc = __builtin_amdgcn_wmma_f32_16x16x32_bf16(false, a, false, b,
                                                      (short)0, acc, false, false);
    }
    int col = rt * 16 + (lane & 15);
    int mbase = (lane & 16) ? 8 : 0;
    float bvc = bv[col];
    #pragma unroll
    for (int r = 0; r < 8; ++r) {
        int grow = gt * 16 + r + mbase;
        outb[(size_t)grow * H + col] = f2bf(acc[r] + bvc);
    }
}

// ---- final = LayerNorm(outb @ Wo^T + bo) * gamma + beta, f32 output.
__global__ __launch_bounds__(256) void k_gemm_o_ln(const unsigned short* __restrict__ outb,
                                                   const unsigned short* __restrict__ wop,
                                                   const float* __restrict__ bo,
                                                   const float* __restrict__ gamma,
                                                   const float* __restrict__ beta,
                                                   float* __restrict__ out) {
    __shared__ float ps[8][16], pq[8][16], mus[16], rsd[16];
    int tid = threadIdx.x, lane = tid & 31, w = tid >> 5;
    int gt = blockIdx.x;

    v8f zero = {};
    v8f acc[4] = { zero, zero, zero, zero };
    #pragma unroll
    for (int kt = 0; kt < 16; ++kt) {
        v16bf a = load_a16(outb, gt * 16, H, kt * 32, lane);
        #pragma unroll
        for (int t = 0; t < 4; ++t) {
            v16bf b = load_b16(wop, w * 4 + t, kt, lane);
            acc[t] = __builtin_amdgcn_wmma_f32_16x16x32_bf16(false, a, false, b,
                                                             (short)0, acc[t], false, false);
        }
    }
    int nsub = lane & 15, mbase = (lane & 16) ? 8 : 0;
    #pragma unroll
    for (int t = 0; t < 4; ++t) {
        int col = (w * 4 + t) * 16 + nsub;
        float b = bo[col];
        #pragma unroll
        for (int r = 0; r < 8; ++r) acc[t][r] += b;
    }
    // per-row partial sums / sums of squares; reduce within each 16-lane half
    float rs[8], rq[8];
    #pragma unroll
    for (int r = 0; r < 8; ++r) {
        float s = 0.f, q = 0.f;
        #pragma unroll
        for (int t = 0; t < 4; ++t) { float v = acc[t][r]; s += v; q += v * v; }
        s += __shfl_xor(s, 1); s += __shfl_xor(s, 2); s += __shfl_xor(s, 4); s += __shfl_xor(s, 8);
        q += __shfl_xor(q, 1); q += __shfl_xor(q, 2); q += __shfl_xor(q, 4); q += __shfl_xor(q, 8);
        rs[r] = s; rq[r] = q;
    }
    if (nsub == 0) {
        #pragma unroll
        for (int r = 0; r < 8; ++r) { ps[w][mbase + r] = rs[r]; pq[w][mbase + r] = rq[r]; }
    }
    __syncthreads();
    if (tid < 16) {
        float S = 0.f, Q = 0.f;
        #pragma unroll
        for (int i = 0; i < 8; ++i) { S += ps[i][tid]; Q += pq[i][tid]; }
        float mu = S * (1.0f / H);
        float var = Q * (1.0f / H) - mu * mu;
        mus[tid] = mu;
        rsd[tid] = rsqrtf(var + LN_EPS);
    }
    __syncthreads();
    #pragma unroll
    for (int t = 0; t < 4; ++t) {
        int col = (w * 4 + t) * 16 + nsub;
        float gmm = gamma[col], bta = beta[col];
        #pragma unroll
        for (int r = 0; r < 8; ++r) {
            int m = mbase + r;
            int grow = gt * 16 + m;
            out[(size_t)grow * H + col] = (acc[t][r] - mus[m]) * rsd[m] * gmm + bta;
        }
    }
}

extern "C" void kernel_launch(void* const* d_in, const int* in_sizes, int n_in,
                              void* d_out, int out_size, void* d_ws, size_t ws_size,
                              hipStream_t stream) {
    const float* x     = (const float*)d_in[0];
    const int*   batch = (const int*)d_in[1];
    const float* Wk    = (const float*)d_in[2];
    const float* bk    = (const float*)d_in[3];
    const float* Wv    = (const float*)d_in[4];
    const float* bv    = (const float*)d_in[5];
    const float* Wo    = (const float*)d_in[6];
    const float* bo    = (const float*)d_in[7];
    const float* toxq  = (const float*)d_in[8];
    const float* gamma = (const float*)d_in[9];
    const float* beta  = (const float*)d_in[10];
    int n = in_sizes[0] / H;

    char* ws = (char*)d_ws;
    float*          wscore = (float*)(ws + 0);                     // 16384 B
    float*          sbias  = (float*)(ws + 16384);                 // 32 B
    int*            start  = (int*)(ws + 16640);                   // 16388 B
    unsigned short* wvp    = (unsigned short*)(ws + 33280);        // 512 KB
    unsigned short* wop    = (unsigned short*)(ws + 33280 + 524288);
    unsigned short* xwn    = (unsigned short*)(ws + 33280 + 1048576);          // 32 MB
    unsigned short* outb   = (unsigned short*)(ws + 33280 + 1048576
                                               + (size_t)GNUM * NH * H * 2);   // 4 MB
    float* out = (float*)d_out;

    k_prep_score<<<17, 256, 0, stream>>>(Wk, bk, toxq, wscore, sbias);
    k_pack_w<<<2 * H * H / 256, 256, 0, stream>>>(Wv, Wo, wvp, wop);
    k_start<<<(GNUM + 1 + 255) / 256, 256, 0, stream>>>(batch, n, start);
    k_attn<<<GNUM, 256, 0, stream>>>(x, start, wscore, sbias, xwn);
    k_gemm_v<<<dim3(GNUM / 16, 32), 32, 0, stream>>>(xwn, wvp, bv, outb);
    k_gemm_o_ln<<<GNUM / 16, 256, 0, stream>>>(outb, wop, bo, gamma, beta, out);
}
